// SelfAttention_14027363189250
// MI455X (gfx1250) — compile-verified, matrix operand order checked by reading, not run on hardware
//
#include <hip/hip_runtime.h>

typedef __attribute__((ext_vector_type(16))) _Float16 v16h;
typedef __attribute__((ext_vector_type(8)))  _Float16 v8h;
typedef __attribute__((ext_vector_type(8)))  float    v8f;

#define NB   8
#define CCH  64
#define HW   4096
#define KP   8
#define WAVES 4            // 4 waves/WG, each wave owns 2 p-tiles (32 rows)

#define LOG2E 1.4426950408889634f

#if __has_builtin(__builtin_amdgcn_exp2f)
#define EXP2F(x) __builtin_amdgcn_exp2f(x)
#else
#define EXP2F(x) exp2f(x)
#endif

#if __has_builtin(__builtin_amdgcn_global_load_async_to_lds_b128) && \
    __has_builtin(__builtin_amdgcn_global_load_async_to_lds_b32)
#define USE_ASYNC_LDS 1
#endif

// exact parameter types per hipcc diagnostic: AS(1)/AS(3) gcc-vector int ptrs
typedef __attribute__((__vector_size__(16))) int i32x4;
#define GLB_I32X4P __attribute__((address_space(1))) i32x4*
#define LDS_I32X4P __attribute__((address_space(3))) i32x4*
#define GLB_I32P   __attribute__((address_space(1))) int*
#define LDS_I32P   __attribute__((address_space(3))) int*

// ---------------- Phase 1: 1x1-conv projections -> packed f16 f/g + f32 s ----
// f is pre-scaled by log2(e) so the attention scores arrive in exp2 domain.
__global__ __launch_bounds__(256) void proj_kernel(
    const float* __restrict__ x,
    const float* __restrict__ Wf, const float* __restrict__ bf,
    const float* __restrict__ Wg, const float* __restrict__ bg,
    const float* __restrict__ Ws, const float* __restrict__ bs,
    v8h* __restrict__ fpk, v8h* __restrict__ gpk, float* __restrict__ sbuf)
{
    __shared__ float sWf[KP * CCH];
    __shared__ float sWg[KP * CCH];
    __shared__ float sWs[CCH];

    const int tid = threadIdx.x;
    sWf[tid] = Wf[tid];
    sWf[tid + 256] = Wf[tid + 256];
    sWg[tid] = Wg[tid];
    sWg[tid + 256] = Wg[tid + 256];
    if (tid < CCH) sWs[tid] = Ws[tid];
    __syncthreads();

    const int pix = blockIdx.x * 256 + tid;          // 0 .. N*HW-1
    const int n = pix >> 12;
    const int p = pix & (HW - 1);
    const float* xp = x + (size_t)n * CCH * HW + p;

    float fa[KP], ga[KP], sa = 0.f;
#pragma unroll
    for (int k = 0; k < KP; ++k) { fa[k] = 0.f; ga[k] = 0.f; }

#pragma unroll 4
    for (int c = 0; c < CCH; ++c) {
        const float xv = xp[(size_t)c * HW];
        sa = fmaf(sWs[c], xv, sa);
#pragma unroll
        for (int k = 0; k < KP; ++k) {
            fa[k] = fmaf(sWf[k * CCH + c], xv, fa[k]);
            ga[k] = fmaf(sWg[k * CCH + c], xv, ga[k]);
        }
    }

    v8h fh, gh;
#pragma unroll
    for (int k = 0; k < KP; ++k) {
        fh[k] = (_Float16)((fa[k] + bf[k]) * LOG2E);  // exp2-domain scores
        gh[k] = (_Float16)(ga[k] + bg[k]);
    }
    fpk[pix] = fh;
    gpk[pix] = gh;
    sbuf[pix] = sa + bs[0];
}

// ---------------- Phase 2: fused WMMA attention + softmax + epilogue ---------
__global__ __launch_bounds__(128) void attn_kernel(
    const v8h* __restrict__ fpk, const v8h* __restrict__ gpk,
    const float* __restrict__ sbuf, const float* __restrict__ x,
    const float* __restrict__ gamma,
    float* __restrict__ out_att, float* __restrict__ out2)
{
    __shared__ v8h   Bsh[HW + 1];     // full batch of g (64KB) + zero slot
    __shared__ float Ssh[HW];         // full batch of s  (16KB)
    __shared__ float att_s[WAVES * 32];

    const int tid  = threadIdx.x;
    const int lane = tid & 31;
    const int wv   = tid >> 5;
    const int n    = blockIdx.x >> 5;                        // batch
    const int p0   = ((blockIdx.x & 31) * WAVES + wv) * 32;  // 32 rows / wave

    // zero slot for lanes >= 16 B-operand reads (covered by the barrier)
    if (tid == 0) {
        v8h z;
#pragma unroll
        for (int i = 0; i < 8; ++i) z[i] = (_Float16)0.f;
        Bsh[HW] = z;
    }

    // ---- stage whole batch of g/s into LDS (async DMA if available) --------
    const v8h*   gsrc = gpk  + (size_t)n * HW;
    const float* ssrc = sbuf + (size_t)n * HW;
#pragma unroll 4
    for (int it = 0; it < HW / 128; ++it) {                  // 32 iters
        const int qi = it * 128 + tid;
#if defined(USE_ASYNC_LDS)
        __builtin_amdgcn_global_load_async_to_lds_b128(
            (GLB_I32X4P)(gsrc + qi), (LDS_I32X4P)&Bsh[qi], 0, 0);
        __builtin_amdgcn_global_load_async_to_lds_b32(
            (GLB_I32P)(ssrc + qi), (LDS_I32P)&Ssh[qi], 0, 0);
#else
        Bsh[qi] = gsrc[qi];
        Ssh[qi] = ssrc[qi];
#endif
    }
#if defined(USE_ASYNC_LDS)
#if __has_builtin(__builtin_amdgcn_s_wait_asynccnt)
    __builtin_amdgcn_s_wait_asynccnt(0);
#else
    asm volatile("s_wait_asynccnt 0" ::: "memory");
#endif
#endif
    asm volatile("" ::: "memory");
    __syncthreads();

    // ---- two A tiles: rows p0..p0+15 and p0+16..p0+31 ----------------------
    v16h a0, a1;
#pragma unroll
    for (int i = 0; i < 16; ++i) { a0[i] = (_Float16)0.f; a1[i] = (_Float16)0.f; }
    if (lane < 16) {
        const v8h f0 = fpk[(size_t)n * HW + p0 + lane];
        const v8h f1 = fpk[(size_t)n * HW + p0 + 16 + lane];
#pragma unroll
        for (int i = 0; i < 8; ++i) { a0[i] = f0[i]; a1[i] = f1[i]; }
    }

    v8h zero8;
#pragma unroll
    for (int i = 0; i < 8; ++i) zero8[i] = (_Float16)0.f;
    v8f cz;
#pragma unroll
    for (int i = 0; i < 8; ++i) cz[i] = 0.f;

    float accl0[8], accw0[8], accl1[8], accw1[8];
#pragma unroll
    for (int r = 0; r < 8; ++r) {
        accl0[r] = 0.f; accw0[r] = 0.f; accl1[r] = 0.f; accw1[r] = 0.f;
    }

    int bidx = (lane < 16) ? lane : HW;        // hi lanes pinned to zero slot
    const int bstep = (lane < 16) ? 16 : 0;
    int sidx = lane & 15;

    for (int t = 0; t < HW / 16; ++t) {        // 256 q-tiles
        const v8h blo = Bsh[bidx];
        const v16h b = __builtin_shufflevector(
            blo, zero8, 0, 1, 2, 3, 4, 5, 6, 7, 8, 9, 10, 11, 12, 13, 14, 15);
        const v8f c0 = __builtin_amdgcn_wmma_f32_16x16x32_f16(
            false, a0, false, b, (short)0, cz, false, false);
        const v8f c1 = __builtin_amdgcn_wmma_f32_16x16x32_f16(
            false, a1, false, b, (short)0, cz, false, false);
        const float sv = Ssh[sidx];
#pragma unroll
        for (int r = 0; r < 8; ++r) {
            const float e0 = EXP2F(c0[r]);     // scores pre-scaled by log2(e)
            accl0[r] += e0;
            accw0[r] = fmaf(e0, sv, accw0[r]);
        }
#pragma unroll
        for (int r = 0; r < 8; ++r) {
            const float e1 = EXP2F(c1[r]);
            accl1[r] += e1;
            accw1[r] = fmaf(e1, sv, accw1[r]);
        }
        bidx += bstep;
        sidx += 16;
    }

    // ---- reduce across the 16 columns held by each half-wave ---------------
    float attv0[8], attv1[8];
#pragma unroll
    for (int r = 0; r < 8; ++r) {
        float l0 = accl0[r], w0 = accw0[r], l1 = accl1[r], w1 = accw1[r];
#pragma unroll
        for (int m = 1; m < 16; m <<= 1) {
            l0 += __shfl_xor(l0, m, 32);
            w0 += __shfl_xor(w0, m, 32);
            l1 += __shfl_xor(l1, m, 32);
            w1 += __shfl_xor(w1, m, 32);
        }
        attv0[r] = w0 / l0;                    // softmax-weighted sum of s
        attv1[r] = w1 / l1;
    }
    if (lane == 0) {
#pragma unroll
        for (int r = 0; r < 8; ++r) {
            att_s[wv * 32 + r]      = attv0[r];   // rows p0   .. p0+7
            att_s[wv * 32 + 16 + r] = attv1[r];   // rows p0+16.. p0+23
        }
    }
    if (lane == 16) {
#pragma unroll
        for (int r = 0; r < 8; ++r) {
            att_s[wv * 32 + 8 + r]  = attv0[r];   // rows p0+8 .. p0+15
            att_s[wv * 32 + 24 + r] = attv1[r];   // rows p0+24.. p0+31
        }
    }
    __syncthreads();

    // ---- outputs -----------------------------------------------------------
    const float ap = att_s[wv * 32 + lane];
    const int   p  = p0 + lane;
    out_att[(size_t)n * HW + p] = ap;

    const float g0 = gamma[0];
#pragma unroll 4
    for (int c = 0; c < CCH; ++c) {
        const size_t idx = (size_t)n * CCH * HW + (size_t)c * HW + p;
        out2[idx] = fmaf(g0, ap, x[idx]);
    }
}

extern "C" void kernel_launch(void* const* d_in, const int* in_sizes, int n_in,
                              void* d_out, int out_size, void* d_ws, size_t ws_size,
                              hipStream_t stream) {
    const float* x     = (const float*)d_in[0];
    const float* Wf    = (const float*)d_in[1];
    const float* bf    = (const float*)d_in[2];
    const float* Wg    = (const float*)d_in[3];
    const float* bg    = (const float*)d_in[4];
    const float* Ws    = (const float*)d_in[5];
    const float* bs    = (const float*)d_in[6];
    const float* gamma = (const float*)d_in[7];

    float* out_att = (float*)d_out;                 // [N, 1, H, W]
    float* out2    = out_att + (size_t)NB * HW;     // [N, C, H, W]

    v8h*   fpk  = (v8h*)d_ws;                       // 512 KB
    v8h*   gpk  = fpk + (size_t)NB * HW;            // 512 KB
    float* sbuf = (float*)(gpk + (size_t)NB * HW);  // 128 KB

    proj_kernel<<<(NB * HW) / 256, 256, 0, stream>>>(
        x, Wf, bf, Wg, bg, Ws, bs, fpk, gpk, sbuf);
    attn_kernel<<<NB * 32, 128, 0, stream>>>(
        fpk, gpk, sbuf, x, gamma, out_att, out2);
}
